// ChamfersDistance_6433861009596
// MI455X (gfx1250) — matrix operand order repaired and checked by reading, not compiled, not run on hardware
//
#include <hip/hip_runtime.h>
#include <hip/hip_bf16.h>

// Problem constants (fixed by the reference setup_inputs()).
#define BB 8
#define NN 4096
#define MM 4096
#define KK 64

typedef __attribute__((ext_vector_type(16))) __bf16 v16bf;
typedef __attribute__((ext_vector_type(8)))  __bf16 v8bf;
typedef __attribute__((ext_vector_type(8)))  float  v8f;

// ---------------------------------------------------------------------------
// Prep: per row of a [rows, 64] f32 matrix: sum of squares, bf16 hi/lo split,
// and init the matching min-buffer entry to +inf bits (valid: values >= 0).
// ---------------------------------------------------------------------------
__global__ void __launch_bounds__(256)
chamfer_prep(const float* __restrict__ in,
             __bf16* __restrict__ hi, __bf16* __restrict__ lo,
             float* __restrict__ sumsq, unsigned int* __restrict__ minbits,
             int rows)
{
    int r = blockIdx.x * blockDim.x + threadIdx.x;
    if (r >= rows) return;
    const float* p  = in + (size_t)r * KK;
    __bf16*      ph = hi + (size_t)r * KK;
    __bf16*      pl = lo + (size_t)r * KK;
    float s = 0.0f;
#pragma unroll
    for (int k = 0; k < KK; ++k) {
        float v = p[k];
        s += v * v;
        __bf16 h = (__bf16)v;           // RNE round to bf16
        float  res = v - (float)h;
        ph[k] = h;
        pl[k] = (__bf16)res;
    }
    sumsq[r]   = s;
    minbits[r] = 0x7f800000u;           // +inf
}

// A-fragment load for V_WMMA_*_16x16x32 (16-bit A, this file's contiguous k order, same as B):
// lanes 0-15 hold row M=lane: K = kc+0..15
// lanes 16-31 hold row M=lane-16: K = kc+16..31
__device__ __forceinline__ v16bf load_a_frag(const __bf16* rowbase, int kc, int laneHi)
{
    const int kb = kc + (laneHi ? 16 : 0);
    v8bf c0 = *(const v8bf*)(rowbase + kb);
    v8bf c1 = *(const v8bf*)(rowbase + kb + 8);
    return __builtin_shufflevector(c0, c1,
                                   0, 1, 2, 3, 4, 5, 6, 7,
                                   8, 9, 10, 11, 12, 13, 14, 15);
}

// One m-tile's worth of B operands + its column norms.
struct BTile {
    v16bf bh0, bh1;   // yhi fragments, K-chunks 0 and 1
    v16bf bl0, bl1;   // ylo fragments
    float y2v;
};

__device__ __forceinline__ void
load_btile(BTile& t, const __bf16* __restrict__ yhi, const __bf16* __restrict__ ylo,
           const float* __restrict__ y2, size_t ybase, int m0, int lane15, int laneHi)
{
    const __bf16* yrh = yhi + (ybase + m0 + lane15) * KK;
    const __bf16* yrl = ylo + (ybase + m0 + lane15) * KK;
    const int ks0 = laneHi * 16;
    const int ks1 = 32 + laneHi * 16;
    t.bh0 = *(const v16bf*)(yrh + ks0);
    t.bh1 = *(const v16bf*)(yrh + ks1);
    t.bl0 = *(const v16bf*)(yrl + ks0);
    t.bl1 = *(const v16bf*)(yrl + ks1);
    t.y2v = y2[ybase + m0 + lane15];
}

// ---------------------------------------------------------------------------
// Main: one block per (b, 32-row N-tile); two 16-row A tiles live in
// registers; 4 waves split the M sweep. Steady state is a 2x-unrolled,
// double-buffered pipeline: tile i+1's loads issue before tile i's 12 WMMAs,
// so L2 latency is hidden behind the previous tile's matrix ops + epilogue.
// Minima are tracked on CLAMPED SQUARED distances (sqrt deferred).
// ---------------------------------------------------------------------------
__global__ void __launch_bounds__(128)
chamfer_main(const __bf16* __restrict__ xhi, const __bf16* __restrict__ xlo,
             const __bf16* __restrict__ yhi, const __bf16* __restrict__ ylo,
             const float* __restrict__ x2,  const float* __restrict__ y2,
             unsigned int* __restrict__ rowminbits,
             unsigned int* __restrict__ colminbits)
{
    const int b      = blockIdx.x / (NN / 32);
    const int n0     = (blockIdx.x % (NN / 32)) * 32;
    const int lane   = threadIdx.x & 31;
    const int wave   = threadIdx.x >> 5;          // 0..3
    const int laneHi = lane >> 4;                 // 0 or 1
    const int lane15 = lane & 15;

    // Preload two 16x64 A tiles (hi and lo): 8 fragments resident.
    v16bf ah[2][2], al[2][2];                     // [rowTile][kChunk]
#pragma unroll
    for (int t = 0; t < 2; ++t) {
        const __bf16* rh = xhi + ((size_t)b * NN + n0 + t * 16 + lane15) * KK;
        const __bf16* rl = xlo + ((size_t)b * NN + n0 + t * 16 + lane15) * KK;
#pragma unroll
        for (int c = 0; c < 2; ++c) {
            ah[t][c] = load_a_frag(rh, c * 32, laneHi);
            al[t][c] = load_a_frag(rl, c * 32, laneHi);
        }
    }

    // C/D layout: VGPR r @ lanes 0-15 -> (row r, col lane); lanes 16-31 -> row r+8.
    float x2r[2][8];
#pragma unroll
    for (int t = 0; t < 2; ++t)
#pragma unroll
        for (int r = 0; r < 8; ++r)
            x2r[t][r] = x2[(size_t)b * NN + n0 + t * 16 + r + laneHi * 8];

    float rmin[2][8];
#pragma unroll
    for (int t = 0; t < 2; ++t)
#pragma unroll
        for (int r = 0; r < 8; ++r) rmin[t][r] = __builtin_inff();

    const size_t ybase = (size_t)b * MM;

    // Process one m-tile: 12 WMMAs over two independent accumulators that
    // share this tile's B fragments, then the squared-distance min epilogue.
    auto process = [&](const BTile& bt, int m0) {
        v8f acc0 = {}, acc1 = {};
        acc0 = __builtin_amdgcn_wmma_f32_16x16x32_bf16(false, ah[0][0], false, bt.bh0, (short)0, acc0, false, false);
        acc1 = __builtin_amdgcn_wmma_f32_16x16x32_bf16(false, ah[1][0], false, bt.bh0, (short)0, acc1, false, false);
        acc0 = __builtin_amdgcn_wmma_f32_16x16x32_bf16(false, al[0][0], false, bt.bh0, (short)0, acc0, false, false);
        acc1 = __builtin_amdgcn_wmma_f32_16x16x32_bf16(false, al[1][0], false, bt.bh0, (short)0, acc1, false, false);
        acc0 = __builtin_amdgcn_wmma_f32_16x16x32_bf16(false, ah[0][0], false, bt.bl0, (short)0, acc0, false, false);
        acc1 = __builtin_amdgcn_wmma_f32_16x16x32_bf16(false, ah[1][0], false, bt.bl0, (short)0, acc1, false, false);
        acc0 = __builtin_amdgcn_wmma_f32_16x16x32_bf16(false, ah[0][1], false, bt.bh1, (short)0, acc0, false, false);
        acc1 = __builtin_amdgcn_wmma_f32_16x16x32_bf16(false, ah[1][1], false, bt.bh1, (short)0, acc1, false, false);
        acc0 = __builtin_amdgcn_wmma_f32_16x16x32_bf16(false, al[0][1], false, bt.bh1, (short)0, acc0, false, false);
        acc1 = __builtin_amdgcn_wmma_f32_16x16x32_bf16(false, al[1][1], false, bt.bh1, (short)0, acc1, false, false);
        acc0 = __builtin_amdgcn_wmma_f32_16x16x32_bf16(false, ah[0][1], false, bt.bl1, (short)0, acc0, false, false);
        acc1 = __builtin_amdgcn_wmma_f32_16x16x32_bf16(false, ah[1][1], false, bt.bl1, (short)0, acc1, false, false);

        float colv = __builtin_inff();
#pragma unroll
        for (int r = 0; r < 8; ++r) {
            float sq0 = fmaxf(fmaf(-2.0f, acc0[r], x2r[0][r] + bt.y2v), 0.0f);
            float sq1 = fmaxf(fmaf(-2.0f, acc1[r], x2r[1][r] + bt.y2v), 0.0f);
            rmin[0][r] = fminf(rmin[0][r], sq0);
            rmin[1][r] = fminf(rmin[1][r], sq1);
            colv = fminf(colv, fminf(sq0, sq1));
        }
        colv = fminf(colv, __shfl_xor(colv, 16, 32));
        if (lane < 16)
            atomicMin(&colminbits[ybase + m0 + lane], __float_as_uint(colv));
    };

    // Double-buffered, 2x-unrolled M sweep: 64 tiles per wave (even count).
    constexpr int NTILES = (MM / 16) / 4;         // per-wave tile count = 64
    BTile buf0, buf1;
    load_btile(buf0, yhi, ylo, y2, ybase, wave * 16, lane15, laneHi);
    for (int k = 0; k < NTILES; k += 2) {
        const int mtA = wave + 4 * k;
        const int mtB = mtA + 4;
        load_btile(buf1, yhi, ylo, y2, ybase, mtB * 16, lane15, laneHi);
        process(buf0, mtA * 16);
        const int mtC = mtA + 8;
        const int mtP = (mtC < MM / 16) ? mtC : wave;   // tail: harmless reload
        load_btile(buf0, yhi, ylo, y2, ybase, mtP * 16, lane15, laneHi);
        process(buf1, mtB * 16);
    }

    // Reduce running row minima (of squared distances) across columns.
#pragma unroll
    for (int t = 0; t < 2; ++t) {
#pragma unroll
        for (int r = 0; r < 8; ++r) {
            float v = rmin[t][r];
            v = fminf(v, __shfl_xor(v, 8, 32));
            v = fminf(v, __shfl_xor(v, 4, 32));
            v = fminf(v, __shfl_xor(v, 2, 32));
            v = fminf(v, __shfl_xor(v, 1, 32));
            if (lane15 == 0)
                atomicMin(&rowminbits[(size_t)b * NN + n0 + t * 16 + r + laneHi * 8],
                          __float_as_uint(v));
        }
    }
}

// ---------------------------------------------------------------------------
// Finalize: buffers hold min squared distances; apply deferred sqrt here.
// loss = mean_b( mean_m sqrt(colmin[b,m]) + mean_n sqrt(rowmin[b,n]) )
// ---------------------------------------------------------------------------
__global__ void __launch_bounds__(256)
chamfer_finalize(const unsigned int* __restrict__ rowminbits,
                 const unsigned int* __restrict__ colminbits,
                 float* __restrict__ out)
{
    __shared__ float sdata[256];
    const int tid = threadIdx.x;
    const float wr = 1.0f / ((float)NN * (float)BB);
    const float wc = 1.0f / ((float)MM * (float)BB);
    float s = 0.0f;
    for (int i = tid; i < BB * NN; i += 256)
        s += sqrtf(__uint_as_float(rowminbits[i])) * wr;
    for (int i = tid; i < BB * MM; i += 256)
        s += sqrtf(__uint_as_float(colminbits[i])) * wc;
    sdata[tid] = s;
    __syncthreads();
    for (int off = 128; off > 0; off >>= 1) {
        if (tid < off) sdata[tid] += sdata[tid + off];
        __syncthreads();
    }
    if (tid == 0) out[0] = sdata[0];
}

extern "C" void kernel_launch(void* const* d_in, const int* in_sizes, int n_in,
                              void* d_out, int out_size, void* d_ws, size_t ws_size,
                              hipStream_t stream)
{
    (void)in_sizes; (void)n_in; (void)out_size; (void)ws_size;
    const float* x = (const float*)d_in[0];   // [B, N, K] f32
    const float* y = (const float*)d_in[1];   // [B, M, K] f32
    float* out = (float*)d_out;

    // Carve workspace (~16.5 MB total), 256B-aligned slices.
    char* ws = (char*)d_ws;
    size_t off = 0;
    auto carve = [&](size_t bytes) -> void* {
        void* p = ws + off;
        off = (off + bytes + 255) & ~(size_t)255;
        return p;
    };
    const size_t nElemsX = (size_t)BB * NN * KK;
    const size_t nElemsY = (size_t)BB * MM * KK;
    __bf16* xhi = (__bf16*)carve(nElemsX * sizeof(__bf16));
    __bf16* xlo = (__bf16*)carve(nElemsX * sizeof(__bf16));
    __bf16* yhi = (__bf16*)carve(nElemsY * sizeof(__bf16));
    __bf16* ylo = (__bf16*)carve(nElemsY * sizeof(__bf16));
    float*  x2  = (float*)carve((size_t)BB * NN * sizeof(float));
    float*  y2  = (float*)carve((size_t)BB * MM * sizeof(float));
    unsigned int* rowminbits = (unsigned int*)carve((size_t)BB * NN * sizeof(unsigned int));
    unsigned int* colminbits = (unsigned int*)carve((size_t)BB * MM * sizeof(unsigned int));

    const int rowsX = BB * NN;
    const int rowsY = BB * MM;
    chamfer_prep<<<(rowsX + 255) / 256, 256, 0, stream>>>(x, xhi, xlo, x2, rowminbits, rowsX);
    chamfer_prep<<<(rowsY + 255) / 256, 256, 0, stream>>>(y, yhi, ylo, y2, colminbits, rowsY);

    chamfer_main<<<BB * (NN / 32), 128, 0, stream>>>(xhi, xlo, yhi, ylo,
                                                     x2, y2, rowminbits, colminbits);

    chamfer_finalize<<<1, 256, 0, stream>>>(rowminbits, colminbits, out);
}
